// fePAM_51642686767659
// MI455X (gfx1250) — matrix operand order repaired and checked by reading, not compiled, not run on hardware
//
#include <hip/hip_runtime.h>
#include <hip/hip_bf16.h>
#include <stdint.h>

// ---------------- problem constants (from reference setup_inputs) ------------
#define BB   4
#define CCH  128          // C == Cv
#define HH   256
#define WW   256
#define HWs  (HH * WW)    // 65536 source positions
#define hh   128
#define wwd  128
#define hwq  (hh * wwd)   // 16384 query pixels
#define KK   16           // keys per pixel

typedef __attribute__((ext_vector_type(16))) __bf16 v16bf;
typedef __attribute__((ext_vector_type(8)))  float  v8f;

struct Pack32 { uint4 lo, hi; };
__device__ __forceinline__ v16bf as_v16bf(uint4 lo, uint4 hi) {
  Pack32 t{lo, hi};
  return __builtin_bit_cast(v16bf, t);
}

// ---------------------------------------------------------------------------
// Kernel 1: transpose (C, N) f32 channel-major -> (N, C) bf16 row-major,
// staged through LDS with CDNA5 async global->LDS loads (ASYNCcnt path).
// ---------------------------------------------------------------------------
__global__ void __launch_bounds__(256)
transpose_pack_bf16(const float* __restrict__ in,
                    __hip_bfloat16* __restrict__ out, int N) {
  __shared__ float tile[32][33];
  const int b     = blockIdx.z;
  const int nbase = blockIdx.x * 32;
  const int cbase = blockIdx.y * 32;
  const int tx = threadIdx.x, ty = threadIdx.y;

  const float* inb = in + (size_t)b * CCH * N;
  __hip_bfloat16* outb = out + (size_t)b * N * CCH;

#pragma unroll
  for (int j = 0; j < 32; j += 8) {
    unsigned lds = (unsigned)(unsigned long long)(void*)&tile[ty + j][tx];
    unsigned long long g =
        (unsigned long long)(const void*)(inb + (size_t)(cbase + ty + j) * N + (nbase + tx));
    // async copy global -> LDS, no VGPR round trip (tracked by ASYNCcnt)
    asm volatile("global_load_async_to_lds_b32 %0, %1, off"
                 :: "v"(lds), "v"(g) : "memory");
  }
  asm volatile("s_wait_asynccnt 0x0" ::: "memory");
  __syncthreads();

#pragma unroll
  for (int j = 0; j < 32; j += 8) {
    float v = tile[tx][ty + j];
    outb[(size_t)(nbase + ty + j) * CCH + (cbase + tx)] = __float2bfloat16(v);
  }
}

// ---------------------------------------------------------------------------
// Kernel 2: one wave32 per 16-pixel tile.
//   phase 1: scores via v_wmma_f32_16x16x32_bf16 (A = Q rows, B = keys of
//            pixel j; keep row j of each D), softmax across lanes,
//   phase 2: out = attn @ Val (VALU, lanes = channels), coalesced stores.
// ---------------------------------------------------------------------------
__global__ void __launch_bounds__(32)
deform_attn_kernel(const int* __restrict__ Pos,
                   const __hip_bfloat16* __restrict__ Qt,
                   const __hip_bfloat16* __restrict__ St,
                   const __hip_bfloat16* __restrict__ Rt,
                   float* __restrict__ OutBuf,
                   float* __restrict__ OutM) {
  __shared__ float attn_s[16][16];
  __shared__ int   fidx[16][16];
  __shared__ float outt[CCH][16];

  const int lane = threadIdx.x;
  const int ln   = lane & 15;
  const int half = lane >> 4;

  const int tile  = blockIdx.x;            // 0 .. B*hw/16-1
  const int b     = tile >> 10;            // hw/16 = 1024 tiles per batch
  const int pbase = (tile & 1023) << 4;

  const char* qtb = (const char*)Qt + ((size_t)b * hwq + pbase) * (CCH * 2);
  const char* stb = (const char*)St + (size_t)b * HWs * (CCH * 2);
  const char* rtb = (const char*)Rt + (size_t)b * HWs * (CCH * 2);

  // ---- A tiles: 16 Q rows x 128 channels (bf16), WMMA A-layout, 4 K-chunks
  v16bf a[4];
  {
    const char* qrow = qtb + (size_t)ln * (CCH * 2);
#pragma unroll
    for (int cc = 0; cc < 4; ++cc) {
      uint4 lo = *(const uint4*)(qrow + cc * 64 + half * 16);       // K 0-7 / 8-15
      uint4 hi = *(const uint4*)(qrow + cc * 64 + 32 + half * 16);  // K 16-23 / 24-31
      a[cc] = as_v16bf(lo, hi);
    }
  }

  const int* posY = Pos + (size_t)(b * 2 + 0) * hwq * KK;
  const int* posX = Pos + (size_t)(b * 2 + 1) * hwq * KK;

  float s[8] = {0, 0, 0, 0, 0, 0, 0, 0};

#pragma unroll
  for (int j = 0; j < 16; ++j) {
    const int p  = pbase + j;
    int iy   = posY[(size_t)p * KK + ln];
    int ix   = posX[(size_t)p * KK + ln];
    int flat = iy * WW + ix;
    if (half == 0) fidx[j][ln] = flat;

    // B tile: the 16 key vectors of pixel j, column n <-> lanes n / n+16
    const char* krow = stb + (size_t)flat * (CCH * 2);
    v8f d = {};
#pragma unroll
    for (int cc = 0; cc < 4; ++cc) {
      uint4 lo = *(const uint4*)(krow + cc * 64 + half * 16);       // K 0-7 / 8-15
      uint4 hi = *(const uint4*)(krow + cc * 64 + 32 + half * 16);  // K 16-23 / 24-31
      v16bf bk = as_v16bf(lo, hi);
      d = __builtin_amdgcn_wmma_f32_16x16x32_bf16(false, a[cc], false, bk,
                                                  (short)0, d, false, false);
    }
    // row j of D is the only valid row; it already sits where row j of the
    // combined score tile lives in C/D layout -> per-lane select, no shuffle.
    const int r = j & 7;
    if ((half == 1) == (j >= 8)) s[r] = d[r];
  }

  // ---- softmax over k (16 lanes per row) + write M output -----------------
#pragma unroll
  for (int r = 0; r < 8; ++r) {
    float m = s[r];
    m = fmaxf(m, __shfl_xor(m, 1));
    m = fmaxf(m, __shfl_xor(m, 2));
    m = fmaxf(m, __shfl_xor(m, 4));
    m = fmaxf(m, __shfl_xor(m, 8));
    float e = __expf(s[r] - m);
    float t = e;
    t += __shfl_xor(t, 1);
    t += __shfl_xor(t, 2);
    t += __shfl_xor(t, 4);
    t += __shfl_xor(t, 8);
    float aw = e / t;
    int Mrow = r + half * 8;
    attn_s[Mrow][ln] = aw;
    OutM[((size_t)b * hwq + pbase + Mrow) * KK + ln] = aw;  // 64B rows, coalesced
  }
  __syncthreads();

  // ---- phase 2: out[p,c] = sum_k attn * Val ; lanes = 4 channels each -----
#pragma unroll 1
  for (int j = 0; j < 16; ++j) {
    float ax = 0.f, ay = 0.f, az = 0.f, awc = 0.f;
#pragma unroll
    for (int kk = 0; kk < 16; ++kk) {
      int   flat = fidx[j][kk];
      float wgt  = attn_s[j][kk];
      const char* vrow = rtb + (size_t)flat * (CCH * 2);
      uint2 pk = *(const uint2*)(vrow + lane * 8);   // 4 bf16 = channels 4L..4L+3
      float v0 = __uint_as_float(pk.x << 16);
      float v1 = __uint_as_float(pk.x & 0xffff0000u);
      float v2 = __uint_as_float(pk.y << 16);
      float v3 = __uint_as_float(pk.y & 0xffff0000u);
      ax  = fmaf(wgt, v0, ax);
      ay  = fmaf(wgt, v1, ay);
      az  = fmaf(wgt, v2, az);
      awc = fmaf(wgt, v3, awc);
    }
    outt[4 * lane + 0][j] = ax;
    outt[4 * lane + 1][j] = ay;
    outt[4 * lane + 2][j] = az;
    outt[4 * lane + 3][j] = awc;
  }
  __syncthreads();

  // ---- coalesced store of (C x 16-pixel) buffer tile ----------------------
#pragma unroll
  for (int rr = 0; rr < 16; ++rr) {
    int c  = rr * 8 + (lane >> 2);
    int p4 = (lane & 3) * 4;
    float4 v = *(const float4*)&outt[c][p4];
    *(float4*)(OutBuf + ((size_t)b * CCH + c) * hwq + pbase + p4) = v;
  }
}

// ---------------------------------------------------------------------------
extern "C" void kernel_launch(void* const* d_in, const int* in_sizes, int n_in,
                              void* d_out, int out_size, void* d_ws, size_t ws_size,
                              hipStream_t stream) {
  const float* Q   = (const float*)d_in[0];
  const float* S   = (const float*)d_in[1];
  const float* R   = (const float*)d_in[2];
  const int*   Pos = (const int*)d_in[3];

  char* ws = (char*)d_ws;
  const size_t srBytes = (size_t)BB * HWs * CCH * 2;  // 64 MiB each
  __hip_bfloat16* St = (__hip_bfloat16*)ws;
  __hip_bfloat16* Rt = (__hip_bfloat16*)(ws + srBytes);
  __hip_bfloat16* Qt = (__hip_bfloat16*)(ws + 2 * srBytes);  // +16 MiB

  dim3 tb(32, 8);
  transpose_pack_bf16<<<dim3(HWs / 32, CCH / 32, BB), tb, 0, stream>>>(S, St, HWs);
  transpose_pack_bf16<<<dim3(HWs / 32, CCH / 32, BB), tb, 0, stream>>>(R, Rt, HWs);
  transpose_pack_bf16<<<dim3(hwq / 32, CCH / 32, BB), tb, 0, stream>>>(Q, Qt, hwq);

  float* OutBuf = (float*)d_out;                       // (B, Cv, h, w)
  float* OutM   = OutBuf + (size_t)BB * CCH * hwq;     // (B, hw, 1, k)

  deform_attn_kernel<<<dim3(BB * hwq / KK), dim3(32), 0, stream>>>(
      Pos, Qt, St, Rt, OutBuf, OutM);
}
